// CRF1_42502996361256
// MI455X (gfx1250) — matrix operand order repaired and checked by reading, not compiled, not run on hardware
//
#include <hip/hip_runtime.h>

typedef __attribute__((ext_vector_type(16))) __bf16 v16bf;
typedef __attribute__((ext_vector_type(8)))  float  v8f;
typedef unsigned short ushort_t;

#define Bn 64
#define Sn 512
#define Dn 1024
#define Hn 1024
#define Tn 9
#define Mn (Bn * Sn)          // 32768 rows

// ---- workspace layout (bytes) ----
#define WS_W1B   0                        // [Hn][Dn] bf16 (W1 transposed) : 2 MB
#define WS_W2B   (2u * 1024u * 1024u)     // [16][Hn] bf16 (W2^T padded)   : 32 KB
#define WS_CRF   (WS_W2B + 32u * 1024u)   // 64 floats
#define WS_HBUF  (4u * 1024u * 1024u)     // [Mn][Hn] bf16                 : 64 MB

__device__ __forceinline__ ushort_t f2bf_rne(float f) {
    unsigned u = __float_as_uint(f);
    u += 0x7FFFu + ((u >> 16) & 1u);      // round-to-nearest-even
    return (ushort_t)(u >> 16);
}
// pack two fp32 -> two bf16 (truncate) in ONE v_perm_b32
__device__ __forceinline__ unsigned pack2t(float lo, float hi) {
    return __builtin_amdgcn_perm(__float_as_uint(hi), __float_as_uint(lo), 0x07060302u);
}

union Frag16 { v16bf v; uint4 q[2]; unsigned u[8]; };

// ------------------------------------------------------------------
// Prep A: W1 [K][N] fp32 -> W1b [N][K] bf16, LDS-tiled transpose
// grid (Dn/32, Hn/32), block (32,8); coalesced reads AND writes
// ------------------------------------------------------------------
__global__ void __launch_bounds__(256) crf_prep_w1(const float* __restrict__ W1,
                                                   ushort_t* __restrict__ W1b) {
    __shared__ ushort_t t[32][33];        // +1 pad: conflict-free transpose
    const int tx = threadIdx.x;           // 0..31
    const int ty = threadIdx.y;           // 0..7
    const int kt = blockIdx.x * 32;       // K tile base
    const int nt = blockIdx.y * 32;       // N tile base
#pragma unroll
    for (int i = 0; i < 4; ++i) {
        int k = ty + 8 * i;
        t[k][tx] = f2bf_rne(W1[(size_t)(kt + k) * Hn + nt + tx]);
    }
    __syncthreads();
#pragma unroll
    for (int i = 0; i < 4; ++i) {
        int n = ty + 8 * i;
        W1b[(size_t)(nt + n) * Dn + kt + tx] = t[tx][n];
    }
}

// Prep B: W2 [K][9] fp32 -> W2b [16][K] bf16 (transposed, zero-padded cols)
__global__ void crf_prep_w2(const float* __restrict__ W2, ushort_t* __restrict__ W2b) {
    int idx = blockIdx.x * 256 + threadIdx.x;     // [16][1024]
    int c = idx >> 10, k = idx & 1023;
    W2b[idx] = f2bf_rne(c < Tn ? W2[k * Tn + c] : 0.0f);
}

// ------------------------------------------------------------------
// GEMM1: h = X @ W1 + b1  (fp32 in, bf16 WMMA, bf16 out to ws)
// grid (Mn/128, Hn/128), 256 threads (8 waves); wave tile 32x64
// (2 A-frags needing fp32->bf16 pack, 4 pure-load B-frags, 8 WMMA)
// ------------------------------------------------------------------
__global__ void __launch_bounds__(256) crf_gemm1(const float* __restrict__ X,
                                                 const ushort_t* __restrict__ W1b,
                                                 const float* __restrict__ b1,
                                                 ushort_t* __restrict__ hbuf) {
    __shared__ ushort_t tile[128 * 128];

    const int lane = threadIdx.x & 31;
    const int wave = threadIdx.x >> 5;
    const int wm = wave >> 1;            // 0..3  (M)
    const int wn = wave & 1;             // 0..1  (N)
    const int lr = lane & 15;
    const int hh = lane >> 4;
    const int bm = blockIdx.x * 128;
    const int bn = blockIdx.y * 128;

    const float* pa[2];
#pragma unroll
    for (int mf = 0; mf < 2; ++mf)
        pa[mf] = X + (size_t)(bm + wm * 32 + mf * 16 + lr) * Dn + 8 * hh;
    const ushort_t* pb[4];
#pragma unroll
    for (int nf = 0; nf < 4; ++nf)
        pb[nf] = W1b + (size_t)(bn + wn * 64 + nf * 16 + lr) * Dn + 8 * hh;

    v8f acc[2][4];
    v8f zero = {};
#pragma unroll
    for (int mf = 0; mf < 2; ++mf)
#pragma unroll
        for (int nf = 0; nf < 4; ++nf) acc[mf][nf] = zero;

#pragma unroll 2
    for (int k = 0; k < Dn; k += 32) {
        Frag16 a[2];
#pragma unroll
        for (int mf = 0; mf < 2; ++mf) {
            const float4* p1 = (const float4*)(pa[mf] + k);        // K = k+8h .. +7
            const float4* p2 = (const float4*)(pa[mf] + k + 16);   // K = k+16+8h .. +7
            float4 f0 = p1[0], f1 = p1[1], f2 = p2[0], f3 = p2[1];
            a[mf].u[0] = pack2t(f0.x, f0.y); a[mf].u[1] = pack2t(f0.z, f0.w);
            a[mf].u[2] = pack2t(f1.x, f1.y); a[mf].u[3] = pack2t(f1.z, f1.w);
            a[mf].u[4] = pack2t(f2.x, f2.y); a[mf].u[5] = pack2t(f2.z, f2.w);
            a[mf].u[6] = pack2t(f3.x, f3.y); a[mf].u[7] = pack2t(f3.z, f3.w);
        }
        Frag16 bfr[4];
#pragma unroll
        for (int nf = 0; nf < 4; ++nf) {
            const uint4* wp = (const uint4*)(pb[nf] + k);
            bfr[nf].q[0] = wp[0];
            bfr[nf].q[1] = wp[2];
        }
#pragma unroll
        for (int mf = 0; mf < 2; ++mf)
#pragma unroll
            for (int nf = 0; nf < 4; ++nf)
                acc[mf][nf] = __builtin_amdgcn_wmma_f32_16x16x32_bf16(
                    false, a[mf].v, false, bfr[nf].v, (short)0, acc[mf][nf], false, false);
    }

    // epilogue: bias + bf16 convert through LDS, then coalesced 128B stores
#pragma unroll
    for (int mf = 0; mf < 2; ++mf)
#pragma unroll
        for (int nf = 0; nf < 4; ++nf) {
            int col = wn * 64 + nf * 16 + lr;
            float bias = b1[bn + col];
#pragma unroll
            for (int v = 0; v < 8; ++v) {
                int row = wm * 32 + mf * 16 + v + 8 * hh;
                tile[row * 128 + col] = f2bf_rne(acc[mf][nf][v] + bias);
            }
        }
    __syncthreads();

    int r = threadIdx.x >> 1;
    int half = threadIdx.x & 1;
    const uint4* src = (const uint4*)(tile + r * 128 + half * 64);
    uint4* dst = (uint4*)(hbuf + (size_t)(bm + r) * Hn + bn + half * 64);
#pragma unroll
    for (int i = 0; i < 8; ++i) dst[i] = src[i];
}

// ------------------------------------------------------------------
// GEMM2: logits = h @ W2 + b2 -> d_out+1 (fp32), N padded 9->16
// grid 256 blocks, 256 threads; one 16x16 C-frag per wave (16 rows)
// ------------------------------------------------------------------
__global__ void __launch_bounds__(256) crf_gemm2(const ushort_t* __restrict__ hbuf,
                                                 const ushort_t* __restrict__ W2b,
                                                 const float* __restrict__ b2,
                                                 float* __restrict__ out) {
    const int lane = threadIdx.x & 31;
    const int wave = threadIdx.x >> 5;
    const int lr = lane & 15;
    const int hh = lane >> 4;
    const int rbase = blockIdx.x * 128 + wave * 16;

    v8f acc = {};
    const ushort_t* hr = hbuf + (size_t)(rbase + lr) * Hn + 8 * hh;
    const ushort_t* wr = W2b + (size_t)lr * Hn + 8 * hh;

#pragma unroll 2
    for (int k = 0; k < Hn; k += 32) {
        Frag16 a, b;
        a.q[0] = ((const uint4*)(hr + k))[0];
        a.q[1] = ((const uint4*)(hr + k + 16))[0];
        const uint4* wp = (const uint4*)(wr + k);
        b.q[0] = wp[0];
        b.q[1] = wp[2];
        acc = __builtin_amdgcn_wmma_f32_16x16x32_bf16(
            false, a.v, false, b.v, (short)0, acc, false, false);
    }

    int N = lr;
    if (N < Tn) {
        float bias = b2[N];
#pragma unroll
        for (int v = 0; v < 8; ++v) {
            int row = rbase + v + 8 * hh;
            out[1 + (size_t)row * Tn + N] = acc[v] + bias;
        }
    }
}

// ------------------------------------------------------------------
// CRF forward: one wave per batch; lanes 0..8 own tag columns
// ------------------------------------------------------------------
__global__ void crf_forward(const float* __restrict__ logits,  // d_out+1
                            const int* __restrict__ labels,
                            const int* __restrict__ mask,
                            const float* __restrict__ startt,
                            const float* __restrict__ endt,
                            const float* __restrict__ trans,
                            float* __restrict__ partial) {
    const int b = blockIdx.x;
    const int j = threadIdx.x;            // 0..31
    const float* em = logits + (size_t)b * Sn * Tn;
    const int* tg = labels + b * Sn;
    const int* mk = mask + b * Sn;

    float tc[Tn];
#pragma unroll
    for (int i = 0; i < Tn; ++i) tc[i] = (j < Tn) ? trans[i * Tn + j] : 0.0f;

    // ---- gold path score (parallel over s, wave-reduced) ----
    float part = 0.0f;
    for (int s = j; s < Sn; s += 32) {
        if (s == 0) {
            part += startt[tg[0]] + em[tg[0]];
        } else {
            float m = (float)mk[s];
            part += (trans[tg[s - 1] * Tn + tg[s]] + em[s * Tn + tg[s]]) * m;
        }
    }
    for (int o = 16; o > 0; o >>= 1) part += __shfl_down(part, o);
    float gold = __shfl(part, 0);
    int lastpos = 0;
    for (int s = 0; s < Sn; ++s)
        if (mk[s] > 0) lastpos = s;
    gold += endt[tg[lastpos]];

    // ---- log-partition forward recurrence ----
    float alpha = (j < Tn) ? (startt[j] + em[j]) : -1e30f;
    for (int s = 1; s < Sn; ++s) {
        float emj = (j < Tn) ? em[s * Tn + j] : 0.0f;
        float t[Tn];
        float m = -1e30f;
#pragma unroll
        for (int i = 0; i < Tn; ++i) {
            float ai = __shfl(alpha, i);
            t[i] = ai + tc[i];
            m = fmaxf(m, t[i]);
        }
        float ssum = 0.0f;
#pragma unroll
        for (int i = 0; i < Tn; ++i) ssum += __expf(t[i] - m);
        float nxt = emj + m + __logf(ssum);
        alpha = (mk[s] > 0) ? nxt : alpha;
    }

    float val = (j < Tn) ? (alpha + endt[j]) : -1e30f;
    float m2 = val;
    for (int o = 16; o > 0; o >>= 1) m2 = fmaxf(m2, __shfl_down(m2, o));
    m2 = __shfl(m2, 0);
    float e = __expf(val - m2);
    for (int o = 16; o > 0; o >>= 1) e += __shfl_down(e, o);
    if (j == 0) partial[b] = (m2 + __logf(e)) - gold;   // log_z - score
}

__global__ void crf_reduce(const float* __restrict__ partial, float* __restrict__ out) {
    if (threadIdx.x == 0) {
        float s = 0.0f;
        for (int i = 0; i < Bn; ++i) s += partial[i];
        out[0] = s * (1.0f / (float)Bn);   // loss = mean(log_z - score)
    }
}

// ------------------------------------------------------------------
extern "C" void kernel_launch(void* const* d_in, const int* in_sizes, int n_in,
                              void* d_out, int out_size, void* d_ws, size_t ws_size,
                              hipStream_t stream) {
    (void)in_sizes; (void)n_in; (void)out_size; (void)ws_size;
    const float* X      = (const float*)d_in[0];
    const int*   labels = (const int*)d_in[1];
    const int*   amask  = (const int*)d_in[2];
    const float* W1     = (const float*)d_in[3];
    const float* b1     = (const float*)d_in[4];
    const float* W2     = (const float*)d_in[5];
    const float* b2     = (const float*)d_in[6];
    const float* startt = (const float*)d_in[7];
    const float* endt   = (const float*)d_in[8];
    const float* trans  = (const float*)d_in[9];
    float* out = (float*)d_out;

    char* ws = (char*)d_ws;
    ushort_t* W1b  = (ushort_t*)(ws + WS_W1B);
    ushort_t* W2b  = (ushort_t*)(ws + WS_W2B);
    float*    crfp = (float*)(ws + WS_CRF);
    ushort_t* hbuf = (ushort_t*)(ws + WS_HBUF);

    // 1) weight conversion / layout (coalesced tiled transpose)
    crf_prep_w1<<<dim3(Dn / 32, Hn / 32), dim3(32, 8), 0, stream>>>(W1, W1b);
    crf_prep_w2<<<(16 * Hn) / 256, 256, 0, stream>>>(W2, W2b);
    // 2) h = X@W1 + b1  (bf16 WMMA)
    crf_gemm1<<<dim3(Mn / 128, Hn / 128), 256, 0, stream>>>(X, W1b, b1, hbuf);
    // 3) logits = h@W2 + b2 -> d_out+1
    crf_gemm2<<<Mn / 128, 256, 0, stream>>>(hbuf, W2b, b2, out);
    // 4) CRF per-batch (log_z - score)
    crf_forward<<<Bn, 32, 0, stream>>>(out + 1, labels, amask, startt, endt, trans, crfp);
    // 5) mean -> loss at d_out[0]
    crf_reduce<<<1, 64, 0, stream>>>(crfp, out);
}
